// MultiConditionModulatedMamba_11725260718319
// MI455X (gfx1250) — compile-verified, matrix operand order checked by reading, not run on hardware
//
#include <hip/hip_runtime.h>
#include <math.h>
#include <stdint.h>

// ---------------- types ----------------
typedef __bf16 bf16;
typedef bf16  v16bf __attribute__((ext_vector_type(16)));
typedef bf16  v8bf  __attribute__((ext_vector_type(8)));
typedef float v8f   __attribute__((ext_vector_type(8)));
typedef float v4f   __attribute__((ext_vector_type(4)));

// exact operand types for the async global->LDS builtins (per clang signature)
typedef int v4i __attribute__((vector_size(16)));
typedef __attribute__((address_space(1))) v4i as1_v4i;   // global, 64-bit ptr
typedef __attribute__((address_space(3))) v4i as3_v4i;   // LDS, 32-bit ptr

// ---------------- problem dims ----------------
#define D_MODEL 1024
#define D_INNER 2048
#define DT_RANK 64
#define D_STATE 64
#define NBATCH  4
#define LSEQ    1024
#define NTOK    (NBATCH*LSEQ)           // 4096 tokens
#define DBL_N   (DT_RANK + 2*D_STATE)   // 192

// ---------------- GEMM tiling ----------------
#define TM 128
#define TN 128
#define TK 32
#define LDSS 40   // bf16 elements per LDS row (32 + 8 pad -> 80B, 16B-aligned chunks)

// CDNA5 async global->LDS path (guarded; falls back to plain loads if absent)
#if defined(__gfx1250__) && __has_builtin(__builtin_amdgcn_global_load_async_to_lds_b128) && __has_builtin(__builtin_amdgcn_s_wait_asynccnt)
#define USE_ASYNC_LDS 1
#else
#define USE_ASYNC_LDS 0
#endif

__device__ __forceinline__ bf16 f2bf(float f) {
  union { float f; unsigned u; } v; v.f = f;
  unsigned short h = (unsigned short)(v.u >> 16);
  bf16 r; __builtin_memcpy(&r, &h, sizeof(r));
  return r;
}
__device__ __forceinline__ float sigmoidf_(float x){ return 1.0f/(1.0f+__expf(-x)); }
__device__ __forceinline__ float siluf_(float x){ return x*sigmoidf_(x); }

#if USE_ASYNC_LDS
// async copy of 16B per lane: global (generic float*) -> LDS (generic float*)
__device__ __forceinline__ void async_b128(const float* g, float* l) {
  __builtin_amdgcn_global_load_async_to_lds_b128(
      (as1_v4i*)(uintptr_t)g,
      (as3_v4i*)(uint32_t)(uintptr_t)l,   // low 32 bits of generic LDS addr = LDS offset
      0, 0);
}
#endif

// Load a 16x32 bf16 fragment (ISA A-layout) from LDS.
// Lane l (<16): row=base+l, K chunks [0,8) and [16,24); lane>=16: K [8,16) and [24,32).
__device__ __forceinline__ v16bf frag_from_lds(const bf16* base, int rowBase, int lane) {
  const int half = lane >> 4;
  const int row  = rowBase + (lane & 15);
  const bf16* p  = base + row * LDSS + half * 8;
  v8bf lo = *(const v8bf*)(p);                  // ds_load_b128
  v8bf hi = *(const v8bf*)(p + 16);             // ds_load_b128
  v16bf out;
#pragma unroll
  for (int i = 0; i < 8; ++i) { out[i] = lo[i]; out[i+8] = hi[i]; }
  return out;
}

// Stage one K-tile: A[128][32] row-major and B transposed into sB[n][k], fp32->bf16.
// Load phase first (8 clauses-able global_load_b128 into regs), then convert+store.
template <bool GUARD>
__device__ __forceinline__ void stage_tile(const float* __restrict__ Ag, int lda,
                                           const float* __restrict__ Bg, int ldb,
                                           int N, int m0, int n0, int kk,
                                           bf16* sA, bf16* sB, int tid)
{
  const int r0 = tid >> 3;            // 0..31
  const int ca = (tid & 7) << 2;      // 0,4,...,28
  const int kq = tid >> 5;            // 0..7
  const int nq = (tid & 31) << 2;     // 0..124

  v4f av[4], bv[4];
#pragma unroll
  for (int p = 0; p < 4; ++p)
    av[p] = *(const v4f*)(Ag + (size_t)(m0 + r0 + p * 32) * lda + kk + ca);
#pragma unroll
  for (int p = 0; p < 4; ++p) {
    if (!GUARD || (n0 + nq + 4 <= N)) {
      bv[p] = *(const v4f*)(Bg + (size_t)(kk + kq + p * 8) * ldb + n0 + nq);
    } else {
#pragma unroll
      for (int i = 0; i < 4; ++i)
        bv[p][i] = (n0 + nq + i < N) ? Bg[(size_t)(kk + kq + p * 8) * ldb + n0 + nq + i]
                                     : 0.0f;
    }
  }

#pragma unroll
  for (int p = 0; p < 4; ++p) {
    bf16* d = sA + (r0 + p * 32) * LDSS + ca;
    d[0] = f2bf(av[p].x); d[1] = f2bf(av[p].y);
    d[2] = f2bf(av[p].z); d[3] = f2bf(av[p].w);
  }
#pragma unroll
  for (int p = 0; p < 4; ++p) {
#pragma unroll
    for (int i = 0; i < 4; ++i)
      sB[(nq + i) * LDSS + (kq + p * 8)] = f2bf(bv[p][i]);
  }
}

// Generic fp32-in / fp32-out GEMM, bf16 WMMA compute, double-buffered LDS.
// C = act(A@B + bias). A: MxK (lda), B: KxN (ldb), C: MxN (ldc).
// ACT: 0=none, 1=softplus. GUARD: N-edge guards (N%128 != 0).
// M multiple of 128, K multiple of 32 (all callers satisfy this).
template <int ACT, bool GUARD>
__global__ void __launch_bounds__(256)
gemm_wmma(const float* __restrict__ Ag, int lda,
          const float* __restrict__ Bg, int ldb,
          const float* __restrict__ bias,
          float* __restrict__ Cg, int ldc,
          int M, int N, int K)
{
  __shared__ __align__(16) bf16 sA[2][TM * LDSS];
  __shared__ __align__(16) bf16 sB[2][TN * LDSS];

  const int tid   = threadIdx.x;
  const int lane  = tid & 31;
  const int wid   = tid >> 5;       // 8 waves
  const int waveM = wid & 1;        // 2 wave-rows  -> 64 M each
  const int waveN = wid >> 1;       // 4 wave-cols  -> 32 N each
  const int m0 = blockIdx.y * TM;
  const int n0 = blockIdx.x * TN;

  v8f acc[4][2];
#pragma unroll
  for (int i = 0; i < 4; ++i)
#pragma unroll
    for (int j = 0; j < 2; ++j)
#pragma unroll
      for (int r = 0; r < 8; ++r) acc[i][j][r] = 0.0f;

  const int nk = K / TK;
  stage_tile<GUARD>(Ag, lda, Bg, ldb, N, m0, n0, 0, sA[0], sB[0], tid);
  __syncthreads();

  for (int t = 0; t < nk; ++t) {
    const int cur = t & 1;
    if (t + 1 < nk)   // stage next tile into the other buffer while computing
      stage_tile<GUARD>(Ag, lda, Bg, ldb, N, m0, n0, (t + 1) * TK,
                        sA[cur ^ 1], sB[cur ^ 1], tid);
    if (t + 2 < nk) { // CDNA5 global_prefetch_b8 two tiles ahead
      __builtin_prefetch(Ag + (size_t)m0 * lda + (t + 2) * TK, 0, 0);
      __builtin_prefetch(Bg + (size_t)((t + 2) * TK) * ldb + n0, 0, 0);
    }

    v16bf af[4], bfm[2];
#pragma unroll
    for (int i = 0; i < 4; ++i) af[i]  = frag_from_lds(sA[cur], waveM * 64 + i * 16, lane);
#pragma unroll
    for (int j = 0; j < 2; ++j) bfm[j] = frag_from_lds(sB[cur], waveN * 32 + j * 16, lane);
#pragma unroll
    for (int i = 0; i < 4; ++i)
#pragma unroll
      for (int j = 0; j < 2; ++j)
        acc[i][j] = __builtin_amdgcn_wmma_f32_16x16x32_bf16(
            false, af[i], false, bfm[j], (short)0, acc[i][j], false, false);
    __syncthreads();
  }

  // ---- store C (ISA f32 C/D layout: VGPR r -> row r + 8*(lane>=16), col lane&15) ----
  const int halfsel = lane >> 4;
  const int cn = lane & 15;
#pragma unroll
  for (int i = 0; i < 4; ++i) {
#pragma unroll
    for (int j = 0; j < 2; ++j) {
      const int col = n0 + waveN * 32 + j * 16 + cn;
      if (!GUARD || col < N) {
        const float bval = bias ? bias[col] : 0.0f;
#pragma unroll
        for (int r = 0; r < 8; ++r) {
          const int row = m0 + waveM * 64 + i * 16 + halfsel * 8 + r;
          float v = acc[i][j][r] + bval;
          if (ACT == 1) v = (v > 20.0f) ? v : __logf(1.0f + __expf(v));
          Cg[(size_t)row * ldc + col] = v;
        }
      }
    }
  }
}

// ---------------- depthwise causal conv(4) + SiLU ----------------
__global__ void __launch_bounds__(256)
conv_silu_kernel(const float* __restrict__ xz, const float* __restrict__ cw,
                 const float* __restrict__ cb, float* __restrict__ xc)
{
  const size_t idx = (size_t)blockIdx.x * 256 + threadIdx.x;
  if (idx >= (size_t)NTOK * D_INNER) return;
  const int d = (int)(idx % D_INNER);
  const size_t tok = idx / D_INNER;
  const int l = (int)(tok % LSEQ);
  float acc = cb[d];
#pragma unroll
  for (int j = 0; j < 4; ++j) {
    const int ls = l - 3 + j;
    if (ls >= 0)
      acc += xz[(tok - l + ls) * (size_t)(2 * D_INNER) + d] * cw[d * 4 + j];
  }
  xc[idx] = siluf_(acc);
}

// ---------------- selective scan: one wave32 per (b,d); lane owns n, n+32 ----------------
// Each wave keeps a private 4-deep LDS ring of the per-token B/C row (512B) filled by
// CDNA5 async global->LDS copies issued 3 tokens ahead (ASYNCcnt, in-order completion).
// No cross-wave barriers: producer == consumer wave.
__global__ void __launch_bounds__(256)
scan_kernel(const float* __restrict__ delta, const float* __restrict__ xc,
            const float* __restrict__ dbl,   const float* __restrict__ xz,
            const float* __restrict__ A_log, const float* __restrict__ Dp,
            float* __restrict__ y, float* __restrict__ state_out)
{
  const int lane = threadIdx.x & 31;
  const int wid  = threadIdx.x >> 5;
  const int b = blockIdx.x >> 8;                 // 256 blocks per batch
  const int d = ((blockIdx.x & 255) << 3) + wid; // 8 channels per block
  const int n0 = lane, n1 = lane + 32;

  const float a0 = -__expf(A_log[d * D_STATE + n0]);
  const float a1 = -__expf(A_log[d * D_STATE + n1]);
  const float dval = Dp[d];
  float s0 = 0.0f, s1 = 0.0f;

  const size_t tok0 = (size_t)b * LSEQ;

#if USE_ASYNC_LDS
  __shared__ __align__(16) float sBC[8][4][128];   // per-wave 4-deep ring (16 KB)
  float (*myBC)[128] = sBC[wid];
#pragma unroll
  for (int p = 0; p < 3; ++p)   // prime the pipeline 3 tokens deep
    async_b128(dbl + (tok0 + p) * DBL_N + DT_RANK + lane * 4, &myBC[p][lane * 4]);
#else
  float bn0n = dbl[tok0 * DBL_N + DT_RANK + n0];
  float bn1n = dbl[tok0 * DBL_N + DT_RANK + n1];
  float cn0n = dbl[tok0 * DBL_N + DT_RANK + D_STATE + n0];
  float cn1n = dbl[tok0 * DBL_N + DT_RANK + D_STATE + n1];
#endif

  // register-pipeline the wave-uniform loads one step ahead
  float dt_n = delta[tok0 * D_INNER + d];
  float u_n  = xc[tok0 * D_INNER + d];

  for (int l = 0; l < LSEQ; ++l) {
    const size_t tok = tok0 + l;
    const float dt = dt_n, u = u_n;
    if (l + 1 < LSEQ) {
      dt_n = delta[(tok + 1) * D_INNER + d];
      u_n  = xc[(tok + 1) * D_INNER + d];
    }

#if USE_ASYNC_LDS
    if (l + 3 < LSEQ) {
      async_b128(dbl + (tok + 3) * DBL_N + DT_RANK + lane * 4, &myBC[(l + 3) & 3][lane * 4]);
      __builtin_amdgcn_s_wait_asynccnt(3);   // oldest (token l) complete; 3 still in flight
    } else if (l + 2 < LSEQ) {
      __builtin_amdgcn_s_wait_asynccnt(2);
    } else if (l + 1 < LSEQ) {
      __builtin_amdgcn_s_wait_asynccnt(1);
    } else {
      __builtin_amdgcn_s_wait_asynccnt(0);
    }
    const float* bc = myBC[l & 3];
    const float bn0 = bc[n0];
    const float bn1 = bc[n1];
    const float cn0 = bc[64 + n0];
    const float cn1 = bc[64 + n1];
#else
    const float bn0 = bn0n, bn1 = bn1n, cn0 = cn0n, cn1 = cn1n;
    if (l + 1 < LSEQ) {
      bn0n = dbl[(tok + 1) * DBL_N + DT_RANK + n0];
      bn1n = dbl[(tok + 1) * DBL_N + DT_RANK + n1];
      cn0n = dbl[(tok + 1) * DBL_N + DT_RANK + D_STATE + n0];
      cn1n = dbl[(tok + 1) * DBL_N + DT_RANK + D_STATE + n1];
    }
#endif

    const float du = dt * u;
    s0 = s0 * __expf(dt * a0) + du * bn0;
    s1 = s1 * __expf(dt * a1) + du * bn1;
    float part = s0 * cn0 + s1 * cn1;
#pragma unroll
    for (int off = 16; off >= 1; off >>= 1)
      part += __shfl_xor(part, off, 32);
    if (lane == 0) {
      const float zz = xz[tok * (size_t)(2 * D_INNER) + D_INNER + d];
      y[tok * D_INNER + d] = (part + u * dval) * siluf_(zz);
    }
  }
  state_out[((size_t)b * D_INNER + d) * D_STATE + n0] = s0;
  state_out[((size_t)b * D_INNER + d) * D_STATE + n1] = s1;
}

// ---------------- conditioning: silu(emo@W_emo+b) + silu(scn@W_scn+b) ----------------
__global__ void __launch_bounds__(256)
cond_kernel(const float* __restrict__ emotion, const float* __restrict__ scene,
            const float* __restrict__ W_emo, const float* __restrict__ b_emo,
            const float* __restrict__ W_scn, const float* __restrict__ b_scn,
            float* __restrict__ cond)
{
  const int idx = blockIdx.x * 256 + threadIdx.x;
  if (idx >= NBATCH * D_MODEL) return;
  const int b = idx / D_MODEL, j = idx % D_MODEL;
  float e = b_emo[j], s = b_scn[j];
  for (int k = 0; k < 256; ++k) {
    e += emotion[b * 256 + k] * W_emo[k * D_MODEL + j];
    s += scene[b * 256 + k]   * W_scn[k * D_MODEL + j];
  }
  cond[idx] = siluf_(e) + siluf_(s);
}

// ---------------- modulation params: cond@W_mod + b_mod ----------------
__global__ void __launch_bounds__(256)
mod_kernel(const float* __restrict__ cond, const float* __restrict__ W_mod,
           const float* __restrict__ b_mod, float* __restrict__ modc)
{
  const int idx = blockIdx.x * 256 + threadIdx.x;
  if (idx >= NBATCH * 2 * D_MODEL) return;
  const int b = idx / (2 * D_MODEL), j = idx % (2 * D_MODEL);
  float a = b_mod[j];
  for (int k = 0; k < D_MODEL; ++k)
    a += cond[b * D_MODEL + k] * W_mod[k * 2 * D_MODEL + j];
  modc[idx] = a;
}

// ---------------- residual + layernorm + FiLM ----------------
__global__ void __launch_bounds__(256)
final_kernel(const float* __restrict__ x, const float* __restrict__ hm,
             const float* __restrict__ ln_g, const float* __restrict__ ln_b,
             const float* __restrict__ modc, float* __restrict__ out)
{
  __shared__ float red[256];
  const int row = blockIdx.x;       // b*L + l
  const int tid = threadIdx.x;
  const int b = row >> 10;
  float v[4]; float sum = 0.f, sq = 0.f;
#pragma unroll
  for (int i = 0; i < 4; ++i) {
    const int j = tid + i * 256;
    const float h = x[(size_t)row * D_MODEL + j] + hm[(size_t)row * D_MODEL + j];
    v[i] = h; sum += h; sq += h * h;
  }
  red[tid] = sum; __syncthreads();
  for (int s = 128; s > 0; s >>= 1) { if (tid < s) red[tid] += red[tid + s]; __syncthreads(); }
  const float mean = red[0] / (float)D_MODEL;
  __syncthreads();
  red[tid] = sq; __syncthreads();
  for (int s = 128; s > 0; s >>= 1) { if (tid < s) red[tid] += red[tid + s]; __syncthreads(); }
  const float var  = red[0] / (float)D_MODEL - mean * mean;
  const float rstd = rsqrtf(var + 1e-5f);
#pragma unroll
  for (int i = 0; i < 4; ++i) {
    const int j = tid + i * 256;
    const float norm = (v[i] - mean) * rstd * ln_g[j] + ln_b[j];
    const float g  = modc[(size_t)b * 2 * D_MODEL + j];
    const float be = modc[(size_t)b * 2 * D_MODEL + D_MODEL + j];
    out[(size_t)row * D_MODEL + j] = norm * (1.0f + g) + be;
  }
}

// ---------------- launcher ----------------
extern "C" void kernel_launch(void* const* d_in, const int* in_sizes, int n_in,
                              void* d_out, int out_size, void* d_ws, size_t ws_size,
                              hipStream_t stream) {
  const float* x       = (const float*)d_in[0];
  const float* emotion = (const float*)d_in[1];
  const float* scene   = (const float*)d_in[2];
  const float* W_in    = (const float*)d_in[3];
  const float* b_in    = (const float*)d_in[4];
  const float* conv_w  = (const float*)d_in[5];
  const float* conv_b  = (const float*)d_in[6];
  const float* W_xproj = (const float*)d_in[7];
  const float* W_dt    = (const float*)d_in[8];
  const float* b_dt    = (const float*)d_in[9];
  const float* A_log   = (const float*)d_in[10];
  const float* D_param = (const float*)d_in[11];
  const float* W_out   = (const float*)d_in[12];
  const float* b_out   = (const float*)d_in[13];
  const float* ln_g    = (const float*)d_in[14];
  const float* ln_b    = (const float*)d_in[15];
  const float* W_emo   = (const float*)d_in[16];
  const float* b_emo   = (const float*)d_in[17];
  const float* W_scn   = (const float*)d_in[18];
  const float* b_scn   = (const float*)d_in[19];
  const float* W_mod   = (const float*)d_in[20];
  const float* b_mod   = (const float*)d_in[21];

  float* out_h     = (float*)d_out;                                   // (B,L,D_MODEL)
  float* out_state = out_h + (size_t)NTOK * D_MODEL;                  // (B,D_INNER,D_STATE)

  float* ws    = (float*)d_ws;
  float* xz    = ws;                                   // NTOK x 4096
  float* xc    = xz    + (size_t)NTOK * 2 * D_INNER;   // NTOK x 2048
  float* dbl   = xc    + (size_t)NTOK * D_INNER;       // NTOK x 192
  float* delta = dbl   + (size_t)NTOK * DBL_N;         // NTOK x 2048
  float* yb    = delta + (size_t)NTOK * D_INNER;       // NTOK x 2048
  float* hm    = yb    + (size_t)NTOK * D_INNER;       // NTOK x 1024
  float* cond  = hm    + (size_t)NTOK * D_MODEL;       // 4 x 1024
  float* modc  = cond  + (size_t)NBATCH * D_MODEL;     // 4 x 2048

  const dim3 blk(256);

  // 1) xz = x @ W_in + b_in                       (4096 x 4096, K=1024)
  gemm_wmma<0, false><<<dim3(2 * D_INNER / TN, NTOK / TM), blk, 0, stream>>>(
      x, D_MODEL, W_in, 2 * D_INNER, b_in, xz, 2 * D_INNER, NTOK, 2 * D_INNER, D_MODEL);

  // 2) xc = silu(causal_conv4(xp) + conv_b)
  conv_silu_kernel<<<(int)(((size_t)NTOK * D_INNER + 255) / 256), blk, 0, stream>>>(
      xz, conv_w, conv_b, xc);

  // 3) dbl = xc @ W_xproj                         (4096 x 192, K=2048)
  gemm_wmma<0, true><<<dim3((DBL_N + TN - 1) / TN, NTOK / TM), blk, 0, stream>>>(
      xc, D_INNER, W_xproj, DBL_N, nullptr, dbl, DBL_N, NTOK, DBL_N, D_INNER);

  // 4) delta = softplus(dt_raw @ W_dt + b_dt)     (4096 x 2048, K=64)
  gemm_wmma<1, false><<<dim3(D_INNER / TN, NTOK / TM), blk, 0, stream>>>(
      dbl, DBL_N, W_dt, D_INNER, b_dt, delta, D_INNER, NTOK, D_INNER, DT_RANK);

  // 5) selective scan + gating; writes y and final state
  scan_kernel<<<NBATCH * (D_INNER / 8), blk, 0, stream>>>(
      delta, xc, dbl, xz, A_log, D_param, yb, out_state);

  // 6) hm = y @ W_out + b_out                     (4096 x 1024, K=2048)
  gemm_wmma<0, false><<<dim3(D_MODEL / TN, NTOK / TM), blk, 0, stream>>>(
      yb, D_INNER, W_out, D_MODEL, b_out, hm, D_MODEL, NTOK, D_MODEL, D_INNER);

  // 7) conditioning + modulation params
  cond_kernel<<<(NBATCH * D_MODEL + 255) / 256, blk, 0, stream>>>(
      emotion, scene, W_emo, b_emo, W_scn, b_scn, cond);
  mod_kernel<<<(NBATCH * 2 * D_MODEL + 255) / 256, blk, 0, stream>>>(
      cond, W_mod, b_mod, modc);

  // 8) h = LN(x + hm)*(1+gamma) + beta
  final_kernel<<<NTOK, blk, 0, stream>>>(x, hm, ln_g, ln_b, modc, out_h);
}
